// LinearCausalEncoderClassification_29858612641824
// MI455X (gfx1250) — compile-verified, matrix operand order checked by reading, not run on hardware
//
#include <hip/hip_runtime.h>
#include <hip/hip_bf16.h>
#include <stdint.h>

// Model dims
#define Bsz 32
#define Tn  512
#define Dn  512
#define Hn  8
#define NLn 6
#define FFn 2048
#define DHn 64
#define Mn  (Bsz*Tn)   // 16384 rows

typedef __attribute__((ext_vector_type(16))) __bf16 v16bf;
typedef __attribute__((ext_vector_type(8)))  float  v8f;
typedef __attribute__((ext_vector_type(4)))  unsigned int v4u;

union FragU { v4u u4[2]; v16bf v; };

__device__ __forceinline__ unsigned short f32_bf16(float f) {
    unsigned u = __float_as_uint(f);
    unsigned r = u + 0x7FFFu + ((u >> 16) & 1u);   // round-to-nearest-even
    return (unsigned short)(r >> 16);
}

// CDNA5 async copy global -> LDS (512B per wave-instruction: 16B per lane).
// lds_byte_off assumes this kernel's static LDS block starts at offset 0.
__device__ __forceinline__ void async_ld_b128(unsigned lds_byte_off, const void* g) {
    asm volatile("global_load_async_to_lds_b128 %0, %1, off"
                 :: "v"(lds_byte_off), "v"(g) : "memory");
}

// ---------------------------------------------------------------------------
// bf16 WMMA GEMM: C[M][N] = A[M][K] * BT[N][K]^T + bias[N]
// Block tile 128x64, BK=32, 256 threads (8 waves, 2x2 WMMA accum each),
// double-buffered async-LDS pipeline.
// ---------------------------------------------------------------------------
__global__ __launch_bounds__(256) void k_gemm_bf16(
    const unsigned short* __restrict__ A,   // [M][K] bf16
    const unsigned short* __restrict__ BT,  // [N][K] bf16 (pre-transposed weight)
    const float* __restrict__ bias,         // [N]
    float* __restrict__ C,                  // [M][N]
    int M, int N, int K)
{
    // A double-buffer: 2 * 128*32 ushort = 16384 B; B double-buffer: 2 * 64*32 = 8192 B
    __shared__ alignas(16) unsigned short smem[2 * 128 * 32 + 2 * 64 * 32];

    // Escape the LDS base so the compiler knows the async-asm writes it and
    // cannot fold subsequent LDS reads to undef.
    asm volatile("" :: "v"(&smem[0]) : "memory");

    const int tid  = threadIdx.x;
    const int wave = tid >> 5;
    const int lane = tid & 31;
    const int wm   = wave >> 1;        // 0..3
    const int wn   = wave & 1;         // 0..1
    const int m0   = blockIdx.x * 128;
    const int n0   = blockIdx.y * 64;

    const v8f vzero = {0.f,0.f,0.f,0.f,0.f,0.f,0.f,0.f};
    v8f acc[2][2] = {{vzero, vzero}, {vzero, vzero}};

    const int KT = K >> 5;

    // Per-wave async-load coordinates (3 async b128 per wave per stage).
    const int arow = wave * 16 + (lane >> 1);        // 128 rows of A tile
    const int acol = (lane & 1) * 16;                // col half
    const unsigned aoff0 = (unsigned)((arow * 32 + acol) * 2);
    const unsigned short* gA = A + (size_t)(m0 + arow) * K + acol;

    const int brow = wave * 8 + (lane >> 2);         // 64 N-rows of BT tile
    const int bcol = (lane & 3) * 8;
    const unsigned boff0 = (unsigned)(16384 + (brow * 32 + bcol) * 2);
    const unsigned short* gB = BT + (size_t)(n0 + brow) * K + bcol;

    // Prologue: issue stage 0 into buffer 0
    async_ld_b128(aoff0,      gA);
    async_ld_b128(aoff0 + 16, gA + 8);
    async_ld_b128(boff0,      gB);

    int buf = 0;
    for (int kt = 0; kt < KT; ++kt) {
        if (kt + 1 < KT) {
            const unsigned ab = (unsigned)((buf ^ 1) * 8192);  // bytes
            const unsigned bb = (unsigned)((buf ^ 1) * 4096);  // bytes
            const unsigned short* ga = gA + (size_t)(kt + 1) * 32;
            const unsigned short* gb = gB + (size_t)(kt + 1) * 32;
            async_ld_b128(aoff0 + ab,      ga);
            async_ld_b128(aoff0 + ab + 16, ga + 8);
            async_ld_b128(boff0 + bb,      gb);
            asm volatile("s_wait_asynccnt 3" ::: "memory");  // stage kt done
        } else {
            asm volatile("s_wait_asynccnt 0" ::: "memory");
        }
        __syncthreads();

        const unsigned short* As = smem + buf * 4096;           // [128][32]
        const unsigned short* Bs = smem + 8192 + buf * 2048;    // [64][32]
        const int lm = lane & 15;
        const int kh = lane >> 4;   // K half select per ISA bf16 A/B layout

        FragU a0, a1, b0, b1;
        {
            const unsigned short* p;
            p = As + (wm * 32 + lm) * 32 + kh * 8;
            a0.u4[0] = *(const v4u*)p;  a0.u4[1] = *(const v4u*)(p + 16);
            p = As + (wm * 32 + 16 + lm) * 32 + kh * 8;
            a1.u4[0] = *(const v4u*)p;  a1.u4[1] = *(const v4u*)(p + 16);
            p = Bs + (wn * 32 + lm) * 32 + kh * 8;
            b0.u4[0] = *(const v4u*)p;  b0.u4[1] = *(const v4u*)(p + 16);
            p = Bs + (wn * 32 + 16 + lm) * 32 + kh * 8;
            b1.u4[0] = *(const v4u*)p;  b1.u4[1] = *(const v4u*)(p + 16);
        }

        acc[0][0] = __builtin_amdgcn_wmma_f32_16x16x32_bf16(false, a0.v, false, b0.v, (short)0, acc[0][0], false, false);
        acc[0][1] = __builtin_amdgcn_wmma_f32_16x16x32_bf16(false, a0.v, false, b1.v, (short)0, acc[0][1], false, false);
        acc[1][0] = __builtin_amdgcn_wmma_f32_16x16x32_bf16(false, a1.v, false, b0.v, (short)0, acc[1][0], false, false);
        acc[1][1] = __builtin_amdgcn_wmma_f32_16x16x32_bf16(false, a1.v, false, b1.v, (short)0, acc[1][1], false, false);

        __syncthreads();
        buf ^= 1;
    }

    // Epilogue: f32 C/D layout = VGPR r: M = r (lanes 0-15) / 8+r (lanes 16-31), N = lane%16
    const int lm = lane & 15;
    const int rbase = (lane < 16) ? 0 : 8;
#pragma unroll
    for (int i = 0; i < 2; ++i)
#pragma unroll
        for (int j = 0; j < 2; ++j) {
            const int col = n0 + wn * 32 + j * 16 + lm;
            const float bv = bias[col];
#pragma unroll
            for (int r = 0; r < 8; ++r) {
                const int row = m0 + wm * 32 + i * 16 + rbase + r;
                C[(size_t)row * N + col] = acc[i][j][r] + bv;
            }
        }
}

// ---------------------------------------------------------------------------
// Causal linear attention: one block per (b,h), 64 threads; state S[64][64]
// held in registers (column j per thread), normalizer via running k-cumsum.
// q,k,v,o all [B,T,H,DH] viewed as [B*T, D].
// ---------------------------------------------------------------------------
__global__ __launch_bounds__(64) void k_attn(
    const float* __restrict__ q, const float* __restrict__ k,
    const float* __restrict__ v, float* __restrict__ o)
{
    const int bh = blockIdx.x;
    const int b = bh / Hn, h = bh % Hn;
    const int j = threadIdx.x;
    __shared__ float sq[DHn], sk[DHn];
    float S[DHn];
    float kc[DHn];
#pragma unroll
    for (int i = 0; i < DHn; ++i) { S[i] = 0.f; kc[i] = 0.f; }

    for (int t = 0; t < Tn; ++t) {
        const size_t base = ((size_t)(b * Tn + t)) * Dn + h * DHn;
        sq[j] = q[base + j];
        sk[j] = k[base + j];
        const float vj = v[base + j];
        __syncthreads();
        float acc = 0.f, den = 0.f;
#pragma unroll
        for (int dk = 0; dk < DHn; ++dk) {
            const float kk = sk[dk];
            S[dk]  += kk * vj;
            kc[dk] += kk;
            const float qq = sq[dk];
            acc += qq * S[dk];
            den += qq * kc[dk];
        }
        o[base + j] = acc / (den + 1e-6f);
        __syncthreads();
    }
}

// ---------------------------------------------------------------------------
// LayerNorm over D=512 (optionally + residual R): one block (128 thr) per row.
// ---------------------------------------------------------------------------
__global__ __launch_bounds__(128) void k_ln(
    const float* __restrict__ X, const float* __restrict__ R,
    const float* __restrict__ g, const float* __restrict__ bta,
    float* __restrict__ Y, int hasR)
{
    const int row = blockIdx.x;
    const int tid = threadIdx.x;
    __shared__ float red[128];

    float4 xv = ((const float4*)(X + (size_t)row * Dn))[tid];
    if (hasR) {
        float4 rv = ((const float4*)(R + (size_t)row * Dn))[tid];
        xv.x += rv.x; xv.y += rv.y; xv.z += rv.z; xv.w += rv.w;
    }
    red[tid] = xv.x + xv.y + xv.z + xv.w;
    __syncthreads();
    for (int off = 64; off > 0; off >>= 1) { if (tid < off) red[tid] += red[tid + off]; __syncthreads(); }
    const float mean = red[0] * (1.0f / Dn);
    __syncthreads();
    const float dx = xv.x - mean, dy = xv.y - mean, dz = xv.z - mean, dw = xv.w - mean;
    red[tid] = dx*dx + dy*dy + dz*dz + dw*dw;
    __syncthreads();
    for (int off = 64; off > 0; off >>= 1) { if (tid < off) red[tid] += red[tid + off]; __syncthreads(); }
    const float rstd = rsqrtf(red[0] * (1.0f / Dn) + 1e-5f);

    float4 gv = ((const float4*)g)[tid];
    float4 bv = ((const float4*)bta)[tid];
    float4 yv;
    yv.x = dx * rstd * gv.x + bv.x;
    yv.y = dy * rstd * gv.y + bv.y;
    yv.z = dz * rstd * gv.z + bv.z;
    yv.w = dw * rstd * gv.w + bv.w;
    ((float4*)(Y + (size_t)row * Dn))[tid] = yv;
}

// --------------------------- elementwise helpers ---------------------------
__global__ void k_conv_w(const float* __restrict__ W, unsigned short* __restrict__ WT, int K, int N) {
    int idx = blockIdx.x * 256 + threadIdx.x;
    if (idx < K * N) {
        int kk = idx / N, n = idx - kk * N;
        WT[(size_t)n * K + kk] = f32_bf16(W[idx]);   // transpose to [N][K]
    }
}
__global__ void k_f32_to_bf16(const float* __restrict__ X, unsigned short* __restrict__ Y, int n) {
    int idx = blockIdx.x * 256 + threadIdx.x;
    if (idx < n) Y[idx] = f32_bf16(X[idx]);
}
__global__ void k_relu_bf16(const float* __restrict__ X, unsigned short* __restrict__ Y, int n) {
    int idx = blockIdx.x * 256 + threadIdx.x;
    if (idx < n) Y[idx] = f32_bf16(fmaxf(X[idx], 0.f));
}
__global__ void k_phi(float* __restrict__ X, int n) {   // elu(x)+1
    int idx = blockIdx.x * 256 + threadIdx.x;
    if (idx < n) { float x = X[idx]; X[idx] = (x > 0.f) ? (x + 1.f) : __expf(x); }
}

// ------------------------- embedding + sinusoidal PE -----------------------
__global__ __launch_bounds__(128) void k_embed(
    const int* __restrict__ tok, const float* __restrict__ emb, float* __restrict__ X)
{
    const int row = blockIdx.x;           // 0..M-1
    const int t = row % Tn;
    const int tk = tok[row];
#pragma unroll
    for (int rep = 0; rep < 4; ++rep) {
        int d = threadIdx.x + rep * 128;
        int pair = d >> 1;
        float freq = __expf(-(9.210340371976184f / (float)Dn) * (float)(2 * pair));
        float ang = (float)t * freq;
        float pe = (d & 1) ? __cosf(ang) : __sinf(ang);
        X[(size_t)row * Dn + d] = emb[(size_t)tk * Dn + d] + pe;
    }
}

// ----------------- causal prefix mean (cumsum/(t+1)) + mask ----------------
__global__ void k_mean(const float* __restrict__ X, float* __restrict__ out) {
    int idx = blockIdx.x * 256 + threadIdx.x;    // B*D
    if (idx < Bsz * Dn) {
        int b = idx / Dn, d = idx % Dn;
        float acc = 0.f;
        for (int t = 0; t < Tn; ++t) {
            size_t p = ((size_t)(b * Tn + t)) * Dn + d;
            acc += X[p];
            out[p] = acc / (float)(t + 1);
        }
    }
}
__global__ void k_mask(float* __restrict__ out) {
    int idx = blockIdx.x * 256 + threadIdx.x;
    if (idx < Bsz * Tn) out[idx] = 1.0f;
}

// ---------------------------------------------------------------------------
extern "C" void kernel_launch(void* const* d_in, const int* in_sizes, int n_in,
                              void* d_out, int out_size, void* d_ws, size_t ws_size,
                              hipStream_t stream)
{
    (void)in_sizes; (void)n_in; (void)out_size; (void)ws_size;
    const int*   tokens = (const int*)  d_in[0];
    const float* emb    = (const float*)d_in[1];
    const float* Wq = (const float*)d_in[2];  const float* bq = (const float*)d_in[3];
    const float* Wk = (const float*)d_in[4];  const float* bk = (const float*)d_in[5];
    const float* Wv = (const float*)d_in[6];  const float* bv = (const float*)d_in[7];
    const float* Wo = (const float*)d_in[8];  const float* bo = (const float*)d_in[9];
    const float* ln1_g = (const float*)d_in[10]; const float* ln1_b = (const float*)d_in[11];
    const float* ln2_g = (const float*)d_in[12]; const float* ln2_b = (const float*)d_in[13];
    const float* W1 = (const float*)d_in[14]; const float* b1 = (const float*)d_in[15];
    const float* W2 = (const float*)d_in[16]; const float* b2 = (const float*)d_in[17];
    const float* lnf_g = (const float*)d_in[18]; const float* lnf_b = (const float*)d_in[19];

    char* ws = (char*)d_ws;
    size_t o = 0;
    float*          xf  = (float*)(ws + o);          o += (size_t)Mn * Dn  * 4;  // activations
    unsigned short* abf = (unsigned short*)(ws + o); o += (size_t)Mn * FFn * 2;  // bf16 A operand
    float*          qf  = (float*)(ws + o);          o += (size_t)Mn * Dn  * 4;
    float*          kf  = (float*)(ws + o);          o += (size_t)Mn * Dn  * 4;
    float*          vf  = (float*)(ws + o);          o += (size_t)Mn * Dn  * 4;
    float*          tf  = (float*)(ws + o);          o += (size_t)Mn * Dn  * 4;
    float*          h1  = (float*)(ws + o);          o += (size_t)Mn * FFn * 4;  // FFN hidden
    unsigned short* wT  = (unsigned short*)(ws + o);                              // bf16 weights [N][K]
    const size_t LWS = 4 * (size_t)Dn * Dn + 2 * (size_t)Dn * FFn;               // elems per layer

    const int nDD = Dn * Dn,  gDD = (nDD + 255) / 256;
    const int nDF = Dn * FFn, gDF = (nDF + 255) / 256;

    // Convert + transpose all weights once per call
    for (int l = 0; l < NLn; ++l) {
        unsigned short* base = wT + (size_t)l * LWS;
        k_conv_w<<<gDD, 256, 0, stream>>>(Wq + (size_t)l * nDD, base,                    Dn, Dn);
        k_conv_w<<<gDD, 256, 0, stream>>>(Wk + (size_t)l * nDD, base + (size_t)nDD,      Dn, Dn);
        k_conv_w<<<gDD, 256, 0, stream>>>(Wv + (size_t)l * nDD, base + 2*(size_t)nDD,    Dn, Dn);
        k_conv_w<<<gDD, 256, 0, stream>>>(Wo + (size_t)l * nDD, base + 3*(size_t)nDD,    Dn, Dn);
        k_conv_w<<<gDF, 256, 0, stream>>>(W1 + (size_t)l * nDF, base + 4*(size_t)nDD,            Dn, FFn);
        k_conv_w<<<gDF, 256, 0, stream>>>(W2 + (size_t)l * nDF, base + 4*(size_t)nDD + (size_t)nDF, FFn, Dn);
    }

    k_embed<<<Mn, 128, 0, stream>>>(tokens, emb, xf);

    const dim3 gD(Mn / 128, Dn / 64);
    const dim3 gF(Mn / 128, FFn / 64);
    const int nMD = Mn * Dn,  gMD = nMD / 256;
    const int nMF = Mn * FFn, gMF = nMF / 256;

    for (int l = 0; l < NLn; ++l) {
        unsigned short* base = wT + (size_t)l * LWS;
        const unsigned short* wqT = base;
        const unsigned short* wkT = base + (size_t)nDD;
        const unsigned short* wvT = base + 2 * (size_t)nDD;
        const unsigned short* woT = base + 3 * (size_t)nDD;
        const unsigned short* w1T = base + 4 * (size_t)nDD;
        const unsigned short* w2T = w1T + (size_t)nDF;

        // Q/K/V projections
        k_f32_to_bf16<<<gMD, 256, 0, stream>>>(xf, abf, nMD);
        k_gemm_bf16<<<gD, 256, 0, stream>>>(abf, wqT, bq + (size_t)l * Dn, qf, Mn, Dn, Dn);
        k_gemm_bf16<<<gD, 256, 0, stream>>>(abf, wkT, bk + (size_t)l * Dn, kf, Mn, Dn, Dn);
        k_gemm_bf16<<<gD, 256, 0, stream>>>(abf, wvT, bv + (size_t)l * Dn, vf, Mn, Dn, Dn);
        k_phi<<<gMD, 256, 0, stream>>>(qf, nMD);
        k_phi<<<gMD, 256, 0, stream>>>(kf, nMD);

        // Causal linear attention -> tf
        k_attn<<<Bsz * Hn, 64, 0, stream>>>(qf, kf, vf, tf);

        // Output projection (-> qf) + residual LN
        k_f32_to_bf16<<<gMD, 256, 0, stream>>>(tf, abf, nMD);
        k_gemm_bf16<<<gD, 256, 0, stream>>>(abf, woT, bo + (size_t)l * Dn, qf, Mn, Dn, Dn);
        k_ln<<<Mn, 128, 0, stream>>>(xf, qf, ln1_g + (size_t)l * Dn, ln1_b + (size_t)l * Dn, xf, 1);

        // FFN
        k_f32_to_bf16<<<gMD, 256, 0, stream>>>(xf, abf, nMD);
        k_gemm_bf16<<<gF, 256, 0, stream>>>(abf, w1T, b1 + (size_t)l * FFn, h1, Mn, FFn, Dn);
        k_relu_bf16<<<gMF, 256, 0, stream>>>(h1, abf, nMF);
        k_gemm_bf16<<<gD, 256, 0, stream>>>(abf, w2T, b2 + (size_t)l * Dn, tf, Mn, Dn, FFn);
        k_ln<<<Mn, 128, 0, stream>>>(xf, tf, ln2_g + (size_t)l * Dn, ln2_b + (size_t)l * Dn, xf, 1);
    }

    // Final LN, causal prefix mean, mask
    k_ln<<<Mn, 128, 0, stream>>>(xf, nullptr, lnf_g, lnf_b, tf, 0);
    float* out = (float*)d_out;
    k_mean<<<(Bsz * Dn + 255) / 256, 256, 0, stream>>>(tf, out);
    k_mask<<<(Bsz * Tn + 255) / 256, 256, 0, stream>>>(out + (size_t)Mn * Dn);
}
